// NeuralSlamMemory_68728066671070
// MI455X (gfx1250) — matrix operand order repaired and checked by reading, not compiled, not run on hardware
//
#include <hip/hip_runtime.h>

// ---------------------------------------------------------------------------
// NeuralSlamMemory on MI455X (gfx1250, wave32, WMMA + TDM)
//
// Dominant stage (patch scatter-add) is a GEMM: M=16(C), K=289(17x17), N=HW
// -> v_wmma_f32_16x16x32_bf16 (bf16 mandatory: weights ~4e-7 underflow fp16
// normals; bf16 keeps f32 exponent range, f32 accumulation exact).
// Weight window staging uses the Tensor Data Mover: a zero-padded bf16 plane
// in workspace makes every block's 24x80 window an unconditional 2D tile
// copy -> tensor_load_to_lds + s_wait_tensorcnt (no bounds VALU, no per-elem
// global/ds pairs). Roofline: ~0.34 GB moved @ 23.3 TB/s => ~15 us target;
// 24 GFLOP of bf16 WMMA is far below the matrix ceiling, so memory-bound.
// ---------------------------------------------------------------------------

#define Cc   16
#define Hh   1536
#define Ww   1536
#define HW   (Hh * Ww)          // 2359296
#define PS   17
#define PS2  289
#define NBLK (HW / 256)         // 9216 pixel blocks

#define PW   (Ww + 16)          // padded weight plane (8 halo each side)
#define PH   (Hh + 16)
#define NPAD ((PH * PW + 255) / 256)

#define LDSW 80                 // 64 cols + 2*8 halo
#define LDSH 24                 // 8 rows + 2*8 halo

typedef __attribute__((ext_vector_type(16))) __bf16          v16bf_t;
typedef __attribute__((ext_vector_type(16))) unsigned short  v16u_t;
typedef __attribute__((ext_vector_type(8)))  float           v8f_t;
typedef __attribute__((ext_vector_type(4)))  unsigned int    v4u_t;
typedef __attribute__((ext_vector_type(8)))  int             v8i_t;
typedef __attribute__((ext_vector_type(4)))  int             v4i_t;

__device__ __forceinline__ unsigned short f32_to_bf16_rne(float f) {
    unsigned int u = __float_as_uint(f);
    unsigned int r = u + 0x7FFFu + ((u >> 16) & 1u);   // round-to-nearest-even
    return (unsigned short)(r >> 16);
}

__device__ __forceinline__ float softplus_f(float x) {
    return fmaxf(x, 0.0f) + log1pf(__expf(-fabsf(x)));
}

// ---------------------------------------------------------------------------
// k0: softmax of 5x5 shift kernel -> ws[2..26]
// ---------------------------------------------------------------------------
__global__ void init_kernel(const float* __restrict__ shift, float* __restrict__ wsf) {
    __shared__ float sh[32];
    int t = threadIdx.x;                     // 32 threads
    float v = (t < 25) ? __expf(shift[t]) : 0.0f;
    sh[t] = v;
    __syncthreads();
    if (t == 0) {
        float s = 0.0f;
        for (int i = 0; i < 25; ++i) s += sh[i];
        sh[31] = s;
    }
    __syncthreads();
    if (t < 25) wsf[2 + t] = sh[t] / sh[31];
}

// ---------------------------------------------------------------------------
// k1: cosine sim vs key, e = exp(softplus(beta)*sim); block partial sums
// ---------------------------------------------------------------------------
__global__ __launch_bounds__(256) void sim_exp_kernel(
    const float* __restrict__ key, const float* __restrict__ beta,
    const float* __restrict__ mem, float* __restrict__ e_out,
    float* __restrict__ partials)
{
    int hw = blockIdx.x * 256 + threadIdx.x;
    float kv[16]; float kn = 0.0f;
#pragma unroll
    for (int c = 0; c < 16; ++c) { kv[c] = key[c]; kn += kv[c] * kv[c]; }
    kn = fmaxf(sqrtf(kn), 1e-8f);
    float dot = 0.0f, mn = 0.0f;
#pragma unroll
    for (int c = 0; c < 16; ++c) {
        float m = mem[(size_t)c * HW + hw];
        dot += m * kv[c]; mn += m * m;
    }
    mn = fmaxf(sqrtf(mn), 1e-8f);
    float sim = dot / (mn * kn);
    float bp  = softplus_f(beta[0]);
    float ev  = __expf(bp * sim);            // |bp*sim| small -> stable w/o max
    e_out[hw] = ev;

    __shared__ float sh[256];
    sh[threadIdx.x] = ev; __syncthreads();
    for (int st = 128; st > 0; st >>= 1) {
        if (threadIdx.x < st) sh[threadIdx.x] += sh[threadIdx.x + st];
        __syncthreads();
    }
    if (threadIdx.x == 0) partials[blockIdx.x] = sh[0];
}

// ---------------------------------------------------------------------------
// deterministic single-block reduction of partial sums
// ---------------------------------------------------------------------------
__global__ __launch_bounds__(256) void reduce_partials(
    const float* __restrict__ part, int n, float* __restrict__ dst)
{
    __shared__ float sh[256];
    float s = 0.0f;
    for (int i = threadIdx.x; i < n; i += 256) s += part[i];
    sh[threadIdx.x] = s; __syncthreads();
    for (int st = 128; st > 0; st >>= 1) {
        if (threadIdx.x < st) sh[threadIdx.x] += sh[threadIdx.x + st];
        __syncthreads();
    }
    if (threadIdx.x == 0) *dst = sh[0];
}

// ---------------------------------------------------------------------------
// k2: w_g = g*softmax + (1-g)*prev
// ---------------------------------------------------------------------------
__global__ __launch_bounds__(256) void combine_kernel(
    const float* __restrict__ e, const float* __restrict__ prevw,
    const float* __restrict__ gate, const float* __restrict__ sumE,
    float* __restrict__ wg)
{
    int hw = blockIdx.x * 256 + threadIdx.x;
    float g  = 1.0f / (1.0f + __expf(-gate[0]));
    float wc = e[hw] / sumE[0];
    wg[hw] = g * wc + (1.0f - g) * prevw[hw];
}

// ---------------------------------------------------------------------------
// k3: 5x5 circular conv with sk, then sharpen power; block partial sums
// ---------------------------------------------------------------------------
__global__ __launch_bounds__(256) void shift_pow_kernel(
    const float* __restrict__ wg, const float* __restrict__ wsf,
    const float* __restrict__ sharpen, float* __restrict__ wp,
    float* __restrict__ partials)
{
    int hw = blockIdx.x * 256 + threadIdx.x;
    int y = hw / Ww, x = hw - y * Ww;
    float skr[25];
#pragma unroll
    for (int i = 0; i < 25; ++i) skr[i] = wsf[2 + i];

    float s = 0.0f;
#pragma unroll
    for (int dy = 0; dy < 5; ++dy) {
        int yy = y + dy - 2;
        if (yy < 0)    yy += Hh;
        if (yy >= Hh)  yy -= Hh;
#pragma unroll
        for (int dx = 0; dx < 5; ++dx) {
            int xx = x + dx - 2;
            if (xx < 0)   xx += Ww;
            if (xx >= Ww) xx -= Ww;
            s += skr[dy * 5 + dx] * wg[yy * Ww + xx];
        }
    }
    float zeta = softplus_f(sharpen[0]) + 1.0f;
    float v = __expf(zeta * __logf(fmaxf(s, 1e-8f)));
    wp[hw] = v;

    __shared__ float sh[256];
    sh[threadIdx.x] = v; __syncthreads();
    for (int st = 128; st > 0; st >>= 1) {
        if (threadIdx.x < st) sh[threadIdx.x] += sh[threadIdx.x + st];
        __syncthreads();
    }
    if (threadIdx.x == 0) partials[blockIdx.x] = sh[0];
}

// ---------------------------------------------------------------------------
// k4: normalize + convert to bf16 into ZERO-PADDED plane (PH x PW):
//     interior (r-8, c-8) in image -> bf16(wp/sum), border -> 0.
//     Padding makes the TDM tile copy in k5 unconditional.
// ---------------------------------------------------------------------------
__global__ __launch_bounds__(256) void normalize_bf16_pad_kernel(
    const float* __restrict__ wp, const float* __restrict__ sumP,
    unsigned short* __restrict__ wpad)
{
    int idx = blockIdx.x * 256 + threadIdx.x;
    if (idx >= PH * PW) return;
    int r = idx / PW, c = idx - r * PW;
    int gy = r - 8, gx = c - 8;
    unsigned short v = 0;
    if (gy >= 0 && gy < Hh && gx >= 0 && gx < Ww) {
        float f = wp[gy * Ww + gx] / fmaxf(sumP[0], 1e-6f);
        v = f32_to_bf16_rne(f);
    }
    wpad[idx] = v;
}

// ---------------------------------------------------------------------------
// k5: out[c,y,x] = mem[c,y,x] + sum_{k,l} patch[c,k,l] * w[y+8-k, x+8-l]
// GEMM per tile: D(16x16) = A(16x32 patch slab) x B(32x16 shifted windows),
// 10 bf16 WMMAs per tile (K=289 padded to 320).
// Block = 256 thr (8 waves) -> 8-row x 64-col strip. The 24x80 bf16 weight
// window is DMA'd into LDS by the Tensor Data Mover (one wave issues the
// descriptor; TENSORcnt + workgroup barrier synchronize).
// ---------------------------------------------------------------------------
__global__ __launch_bounds__(256) void conv_wmma_kernel(
    const unsigned short* __restrict__ wpad, const float* __restrict__ patch,
    const float* __restrict__ mem, float* __restrict__ out)
{
    __shared__ unsigned short wl[LDSH * LDSW];

    const int tid = threadIdx.x;
    const int x0  = blockIdx.x * 64;
    const int y0  = blockIdx.y * 8;

    const int lane    = tid & 31;
    const int dy      = tid >> 5;      // wave id = output row in strip
    const int nIdx    = lane & 15;     // N column within tile
    const int halfSel = lane >> 4;     // lane half selects K sub-range

    // ---- TDM: 2D tile (80 x 24 elements, 2B each) from padded plane -> LDS.
    // Padded coords: tile origin (row=y0, col=x0) covers global rows y0-8..
    // y0+15, cols x0-8..x0+71 with the +8 pad shift. Fully in-bounds.
    if (dy == 0) {                      // wave 0 issues the DMA (EXEC ignored)
        unsigned long long gaddr =
            (unsigned long long)(uintptr_t)(wpad + (size_t)y0 * PW + x0);
        unsigned int lds_off = (unsigned int)(uintptr_t)(&wl[0]);

        v4u_t g0;
        g0[0] = 1u;                                    // count=1, user desc
        g0[1] = lds_off;                               // lds_addr (bytes)
        g0[2] = (unsigned int)(gaddr & 0xFFFFFFFFull); // global_addr[31:0]
        g0[3] = (unsigned int)((gaddr >> 32) & 0x1FFFFFFull) | (2u << 30); // [56:32] | type=2

        v8i_t g1;
        g1[0] = (int)(1u << 16);                       // data_size=1 (2 bytes)
        g1[1] = (int)(((unsigned)PW & 0xFFFFu) << 16); // tensor_dim0[15:0]
        g1[2] = (int)((((unsigned)PW >> 16) & 0xFFFFu) |
                      (((unsigned)PH & 0xFFFFu) << 16));      // dim0 hi | dim1 lo
        g1[3] = (int)((((unsigned)PH >> 16) & 0xFFFFu) |
                      ((unsigned)LDSW << 16));         // dim1 hi | tile_dim0=80
        g1[4] = (int)LDSH;                             // tile_dim1=24, tile_dim2=0
        g1[5] = (int)PW;                               // tensor_dim0_stride[31:0]
        g1[6] = 0;                                     // stride hi | dim1_stride lo
        g1[7] = 0;

        v4i_t gz4 = {0, 0, 0, 0};
        v8i_t gz8 = {0, 0, 0, 0, 0, 0, 0, 0};
        __builtin_amdgcn_tensor_load_to_lds(g0, g1, gz4, gz4, gz8, 0);
        __builtin_amdgcn_s_wait_tensorcnt(0);
    }

    // A fragments: 16-bit A 16x32 layout. lanes0-15: e<8 -> K=e, e>=8 -> K=e+8
    // lanes16-31: +8. Constant across all tiles; kept in VGPRs (80 regs).
    v16bf_t afrag[10];
#pragma unroll
    for (int t = 0; t < 10; ++t) {
        v16u_t au;
#pragma unroll
        for (int e = 0; e < 16; ++e) {
            int K = 32 * t + (halfSel ? 8 : 0) + (e < 8 ? e : e + 8);
            float pv = (K < PS2) ? patch[nIdx * PS2 + K] : 0.0f;
            au[e] = f32_to_bf16_rne(pv);
        }
        afrag[t] = __builtin_bit_cast(v16bf_t, au);
    }

    __syncthreads();

    // lds addr = laneBase - (k*LDSW + l), kk = k*17+l
    const int laneBase = (dy + 16) * LDSW + 16 + nIdx;

#pragma unroll
    for (int xt = 0; xt < 4; ++xt) {
        const int base = laneBase + xt * 16;
        const int y = y0 + dy;
        const int x = x0 + xt * 16 + nIdx;

        // hoist mem loads so they overlap the WMMA chain
        float mvals[8];
#pragma unroll
        for (int j = 0; j < 8; ++j) {
            int c = j + (halfSel ? 8 : 0);
            mvals[j] = mem[(size_t)c * HW + (size_t)y * Ww + x];
        }

        v8f_t acc = {0.f, 0.f, 0.f, 0.f, 0.f, 0.f, 0.f, 0.f};

#pragma unroll
        for (int t = 0; t < 9; ++t) {          // kk in [32t, 32t+32) all < 289
            v16u_t bu;
#pragma unroll
            for (int e = 0; e < 16; ++e) {
                const int kk0  = 32 * t + (e < 8 ? e : e + 8); // lanes 0-15
                const int kk1  = kk0 + 8;           // lanes 16-31
                const int off0 = (kk0 / PS) * LDSW + (kk0 % PS); // immediates
                const int off1 = (kk1 / PS) * LDSW + (kk1 % PS);
                int off = halfSel ? off1 : off0;
                bu[e] = wl[base - off];
            }
            v16bf_t bfrag = __builtin_bit_cast(v16bf_t, bu);
            acc = __builtin_amdgcn_wmma_f32_16x16x32_bf16(
                false, afrag[t], false, bfrag, (short)0, acc, false, false);
        }
        {   // chunk 9: only kk=288 (k=16,l=16) valid, lanes 0-15 element 0
            v16u_t bu;
#pragma unroll
            for (int e = 1; e < 16; ++e) bu[e] = 0;
            unsigned short v288 = wl[base - (16 * LDSW + 16)];
            bu[0] = halfSel ? (unsigned short)0 : v288;
            v16bf_t bfrag = __builtin_bit_cast(v16bf_t, bu);
            acc = __builtin_amdgcn_wmma_f32_16x16x32_bf16(
                false, afrag[9], false, bfrag, (short)0, acc, false, false);
        }

        // D layout: lanes0-15 vgpr j -> (M=j, N=lane); lanes16-31 -> M=j+8
#pragma unroll
        for (int j = 0; j < 8; ++j) {
            int c = j + (halfSel ? 8 : 0);
            size_t idx = (size_t)c * HW + (size_t)y * Ww + x;
            out[idx] = mvals[j] + acc[j];
        }
    }
}

// ---------------------------------------------------------------------------
extern "C" void kernel_launch(void* const* d_in, const int* in_sizes, int n_in,
                              void* d_out, int out_size, void* d_ws, size_t ws_size,
                              hipStream_t stream) {
    const float* key     = (const float*)d_in[0];
    const float* beta    = (const float*)d_in[1];
    const float* gate    = (const float*)d_in[2];
    const float* shift   = (const float*)d_in[3];
    const float* sharpen = (const float*)d_in[4];
    const float* prevw   = (const float*)d_in[5];
    const float* patch   = (const float*)d_in[6];
    const float* mem     = (const float*)d_in[7];
    float* out = (float*)d_out;

    // ws layout: [0]=sumE [1]=sumP [2..26]=sk | partE[9216] | partP[9216]
    //            | padded bf16 plane (PH*PW ushorts, ~4.8 MB)
    float* wsf   = (float*)d_ws;
    float* partE = wsf + 32;
    float* partP = partE + NBLK;
    unsigned short* wpad = (unsigned short*)(partP + NBLK);

    // d_out planes 0/1 double as f32 scratch; fully rewritten by conv_wmma.
    float* bufA = out;            // e -> w_pow
    float* bufB = out + HW;       // w_g

    init_kernel<<<1, 32, 0, stream>>>(shift, wsf);
    sim_exp_kernel<<<NBLK, 256, 0, stream>>>(key, beta, mem, bufA, partE);
    reduce_partials<<<1, 256, 0, stream>>>(partE, NBLK, wsf + 0);
    combine_kernel<<<NBLK, 256, 0, stream>>>(bufA, prevw, gate, wsf + 0, bufB);
    shift_pow_kernel<<<NBLK, 256, 0, stream>>>(bufB, wsf, sharpen, bufA, partP);
    reduce_partials<<<1, 256, 0, stream>>>(partP, NBLK, wsf + 1);
    normalize_bf16_pad_kernel<<<NPAD, 256, 0, stream>>>(bufA, wsf + 1, wpad);

    dim3 grid(Ww / 64, Hh / 8);   // 24 x 192
    conv_wmma_kernel<<<grid, 256, 0, stream>>>(wpad, patch, mem, out);
}
